// CausalSelfAttention_90245852823577
// MI455X (gfx1250) — matrix operand order repaired and checked by reading, not compiled, not run on hardware
//
#include <hip/hip_runtime.h>
#include <hip/hip_bf16.h>

typedef __bf16 bf16;
typedef __attribute__((ext_vector_type(16))) __bf16 v16bf;
typedef __attribute__((ext_vector_type(8)))  __bf16 v8bf;
typedef __attribute__((ext_vector_type(4)))  __bf16 v4bf;
typedef __attribute__((ext_vector_type(8)))  float  v8f;

#define WMMA_BF16(a, b, c) \
  __builtin_amdgcn_wmma_f32_16x16x32_bf16(false, (a), false, (b), (short)0, (c), false, false)

static __device__ inline v16bf ld_frag(const bf16* p0, const bf16* p1) {
  union { v16bf v; v8bf h[2]; } u;
  u.h[0] = *(const v8bf*)p0;
  u.h[1] = *(const v8bf*)p1;
  return u.v;
}

// wave-local LDS store->load ordering (P staging is per-wave; LDS ops of one
// wave execute in order, we just need the DScnt wait + no compiler reordering)
static __device__ inline void lds_wave_fence() {
  asm volatile("s_wait_dscnt 0" ::: "memory");
}

// gfx1250 async global->LDS copy (GLOBAL_LOAD_ASYNC_TO_LDS_B128, ASYNCcnt).
// VDST = per-lane LDS byte address (low 32 bits of a shared pointer are the
// wave-relative LDS offset per the aperture mapping), VADDR = 64-bit global VA.
static __device__ inline void async_load_b128(const void* gptr, void* lptr) {
  uint32_t lds = (uint32_t)(uintptr_t)lptr;
  uint64_t gva = (uint64_t)(uintptr_t)gptr;
  asm volatile("global_load_async_to_lds_b128 %0, %1, off"
               :: "v"(lds), "v"(gva) : "memory");
}
static __device__ inline void wait_async() {
  asm volatile("s_wait_asynccnt 0" ::: "memory");
}

// ---------------- problem constants ----------------
constexpr int Bb = 4, T = 2048, Cc = 1024, H = 16, DH = 64;
constexpr int M = Bb * T;   // 8192
constexpr int K = Cc;       // 1024
constexpr int N = Cc;       // 1024

// ---------------- fp32 -> bf16 convert ----------------
__global__ __launch_bounds__(256) void cvt_bf16(const float* __restrict__ src,
                                                bf16* __restrict__ dst) {
  int i = (blockIdx.x * 256 + threadIdx.x) * 4;
  float4 f = *(const float4*)(src + i);
  v4bf o;
  o[0] = (bf16)f.x; o[1] = (bf16)f.y; o[2] = (bf16)f.z; o[3] = (bf16)f.w;
  *(v4bf*)(dst + i) = o;
}

// ---------------- bf16 GEMM: out = scale * (A[M,K] @ W[N,K]^T) ----------------
// MODE 0: fp32 out[m*N+n]
// MODE 1: bf16 out[((b*H+h)*T+t)*DH+dh]          (Q, K head layout)
// MODE 2: bf16 out[((b*H+h)*DH+dh)*T+t]          (V transposed)
// Tiles: BM=128, BN=128, BK=64; 8 waves (4x2), each 32x64;
// double-buffered LDS filled by async global->LDS DMA loads.
constexpr int LSA = 72;  // LDS row stride (halfs), 16B aligned, conflict-padded
template <int MODE>
__global__ __launch_bounds__(256) void gemm_bf16(const bf16* __restrict__ A,
                                                 const bf16* __restrict__ W,
                                                 void* __restrict__ out,
                                                 float scale) {
  __shared__ bf16 As[2][128 * LSA];
  __shared__ bf16 Bs[2][128 * LSA];
  const int tid = threadIdx.x;
  const int lane = tid & 31, wid = tid >> 5;
  const int wm = wid & 3, wn = wid >> 2;        // 4x2 wave grid, 32x64 per wave
  const int m0 = blockIdx.x * 128;
  const int n0 = blockIdx.y * 128;
  const int l15 = lane & 15, lhi = lane >> 4;

  v8f acc[2][4] = {};

  // tile fill: both tiles are 128 rows x 64 halfs; each thread owns 32 halfs
  const int srow = tid >> 1, scol = (tid & 1) * 32;
  const bf16* gA = A + (size_t)(m0 + srow) * K + scol;
  const bf16* gW = W + (size_t)(n0 + srow) * K + scol;

  auto load_tile_async = [&](int kt, int buf) {
    bf16* as = As[buf] + srow * LSA + scol;
    bf16* bs = Bs[buf] + srow * LSA + scol;
    async_load_b128(gA + kt,      as);
    async_load_b128(gA + kt + 8,  as + 8);
    async_load_b128(gA + kt + 16, as + 16);
    async_load_b128(gA + kt + 24, as + 24);
    async_load_b128(gW + kt,      bs);
    async_load_b128(gW + kt + 8,  bs + 8);
    async_load_b128(gW + kt + 16, bs + 16);
    async_load_b128(gW + kt + 24, bs + 24);
  };

  load_tile_async(0, 0);
  wait_async();
  __syncthreads();

  constexpr int NIT = K / 64;  // 16
  for (int it = 0; it < NIT; ++it) {
    const int buf = it & 1;
    if (it + 1 < NIT) load_tile_async((it + 1) * 64, buf ^ 1);
#pragma unroll
    for (int kk = 0; kk < 64; kk += 32) {
      // hoist all fragments for this k-slice, then burst the WMMAs
      v16bf af[2], bfr[4];
      for (int mi = 0; mi < 2; ++mi) {
        const bf16* p = As[buf] + (wm * 32 + mi * 16 + l15) * LSA + kk + lhi * 8;
        af[mi] = ld_frag(p, p + 16);
      }
      for (int ni = 0; ni < 4; ++ni) {
        const bf16* p = Bs[buf] + (wn * 64 + ni * 16 + l15) * LSA + kk + lhi * 8;
        bfr[ni] = ld_frag(p, p + 16);
      }
      for (int mi = 0; mi < 2; ++mi)
        for (int ni = 0; ni < 4; ++ni)
          acc[mi][ni] = WMMA_BF16(af[mi], bfr[ni], acc[mi][ni]);
    }
    if (it + 1 < NIT) wait_async();
    __syncthreads();
  }

  for (int mi = 0; mi < 2; ++mi)
    for (int ni = 0; ni < 4; ++ni)
      for (int j = 0; j < 8; ++j) {
        int m = m0 + wm * 32 + mi * 16 + j + lhi * 8;
        int n = n0 + wn * 64 + ni * 16 + l15;
        float v = acc[mi][ni][j] * scale;
        if constexpr (MODE == 0) {
          ((float*)out)[(size_t)m * N + n] = v;
        } else {
          int b = m >> 11, t = m & 2047;
          int h = n >> 6, dh = n & 63;
          if constexpr (MODE == 1)
            ((bf16*)out)[(((size_t)(b * H + h)) * T + t) * DH + dh] = (bf16)v;
          else
            ((bf16*)out)[(((size_t)(b * H + h)) * DH + dh) * T + t] = (bf16)v;
        }
      }
}

// ---------------- flash attention: 128 queries per block, one (b,h) ----------------
// Q is pre-scaled by 1/sqrt(DH) in its projection GEMM.
__global__ __launch_bounds__(256) void attn(const bf16* __restrict__ Q,
                                            const bf16* __restrict__ Kg,
                                            const bf16* __restrict__ Vt,
                                            bf16* __restrict__ O) {
  __shared__ bf16 Ks[2][32 * 72];      // 32 keys x 64 d (padded)
  __shared__ bf16 Vs[2][64 * 40];      // 64 dh x 32 keys (padded)
  __shared__ bf16 Ps[8 * 16 * 40];     // per-wave P staging
  const int tid = threadIdx.x, lane = tid & 31, wid = tid >> 5;
  const int l15 = lane & 15, lhi = lane >> 4;
  const int bh = blockIdx.y;                 // b*H + h
  const int qt0 = blockIdx.x * 128;
  const int qrow0 = qt0 + wid * 16;          // this wave's 16 query rows

  const bf16* Qb = Q  + (size_t)bh * T * DH;
  const bf16* Kb = Kg + (size_t)bh * T * DH;
  const bf16* Vb = Vt + (size_t)bh * DH * T;

  v16bf qf[2];
  for (int d = 0; d < 2; ++d) {
    const bf16* p = Qb + (size_t)(qrow0 + l15) * DH + d * 32 + lhi * 8;
    qf[d] = ld_frag(p, p + 16);
  }

  v8f oacc[4] = {};
  float mrow[8], lrow[8];
  for (int j = 0; j < 8; ++j) { mrow[j] = -1e30f; lrow[j] = 0.f; }

  bf16* Pw = Ps + wid * (16 * 40);
  const int nsteps = qt0 / 32 + 4;           // causal bound for this tile

  const int krow = tid >> 3, kpart = (tid & 7) * 8;  // K tile: 32 rows x 64 halfs
  const int vrow = tid >> 2, vpart = (tid & 3) * 8;  // V tile: 64 rows x 32 halfs
  const bf16* gK = Kb + (size_t)krow * DH + kpart;
  const bf16* gV = Vb + (size_t)vrow * T + vpart;
  bf16* lK0 = Ks[0] + krow * 72 + kpart;
  bf16* lK1 = Ks[1] + krow * 72 + kpart;
  bf16* lV0 = Vs[0] + vrow * 40 + vpart;
  bf16* lV1 = Vs[1] + vrow * 40 + vpart;

  async_load_b128(gK, lK0);
  async_load_b128(gV, lV0);
  wait_async();
  __syncthreads();

  for (int s = 0; s < nsteps; ++s) {
    const int buf = s & 1;
    if (s + 1 < nsteps) {
      async_load_b128(gK + (size_t)(s + 1) * 32 * DH, buf ? lK0 : lK1);
      async_load_b128(gV + (s + 1) * 32,              buf ? lV0 : lV1);
    }
    const int k0 = s * 32;

    // hoist all K fragments, then burst S = Q K^T
    v16bf kf[2][2];
    for (int c = 0; c < 2; ++c)
      for (int d = 0; d < 2; ++d) {
        const bf16* p = Ks[buf] + (c * 16 + l15) * 72 + d * 32 + lhi * 8;
        kf[c][d] = ld_frag(p, p + 16);
      }
    v8f sf[2] = {};
    for (int c = 0; c < 2; ++c)
      for (int d = 0; d < 2; ++d)
        sf[c] = WMMA_BF16(qf[d], kf[c][d], sf[c]);

    // causal mask only where the 32-key window crosses this wave's diagonal
    if (k0 + 31 > qrow0) {
      for (int c = 0; c < 2; ++c)
        for (int j = 0; j < 8; ++j) {
          int q = qrow0 + j + lhi * 8;
          int kk = k0 + c * 16 + l15;
          if (kk > q) sf[c][j] = -1e30f;
        }
    }

    // online softmax (rows owned per-wave; reduce within 16-lane halves)
    for (int j = 0; j < 8; ++j) {
      float mx = fmaxf(sf[0][j], sf[1][j]);
      mx = fmaxf(mx, __shfl_xor(mx, 1));
      mx = fmaxf(mx, __shfl_xor(mx, 2));
      mx = fmaxf(mx, __shfl_xor(mx, 4));
      mx = fmaxf(mx, __shfl_xor(mx, 8));
      float mnew = fmaxf(mrow[j], mx);
      float corr = __expf(mrow[j] - mnew);
      mrow[j] = mnew;
      float p0 = __expf(sf[0][j] - mnew);
      float p1 = __expf(sf[1][j] - mnew);
      sf[0][j] = p0; sf[1][j] = p1;
      float rs = p0 + p1;
      rs += __shfl_xor(rs, 1);
      rs += __shfl_xor(rs, 2);
      rs += __shfl_xor(rs, 4);
      rs += __shfl_xor(rs, 8);
      lrow[j] = lrow[j] * corr + rs;
      for (int f = 0; f < 4; ++f) oacc[f][j] *= corr;
    }

    // C-layout P -> per-wave LDS -> A-layout fragment (wave-local, no barrier)
    for (int c = 0; c < 2; ++c)
      for (int j = 0; j < 8; ++j)
        Pw[(j + lhi * 8) * 40 + c * 16 + l15] = (bf16)sf[c][j];
    lds_wave_fence();
    const bf16* pp = Pw + l15 * 40 + lhi * 8;
    v16bf pf = ld_frag(pp, pp + 16);

    // hoist all V fragments, then burst O += P V
    v16bf vf[4];
    for (int f = 0; f < 4; ++f) {
      const bf16* vp = Vs[buf] + (f * 16 + l15) * 40 + lhi * 8;
      vf[f] = ld_frag(vp, vp + 16);
    }
    for (int f = 0; f < 4; ++f)
      oacc[f] = WMMA_BF16(pf, vf[f], oacc[f]);

    if (s + 1 < nsteps) wait_async();
    __syncthreads();
  }

  // write attention output as bf16 row-major [B*T, C] (input to Wo GEMM)
  const int b = bh >> 4, h = bh & 15;
  for (int f = 0; f < 4; ++f)
    for (int j = 0; j < 8; ++j) {
      int q = qrow0 + j + lhi * 8;
      int cc = h * 64 + f * 16 + l15;
      O[((size_t)(b * T + q)) * Cc + cc] = (bf16)(oacc[f][j] / lrow[j]);
    }
}

// ---------------- host launcher ----------------
extern "C" void kernel_launch(void* const* d_in, const int* in_sizes, int n_in,
                              void* d_out, int out_size, void* d_ws, size_t ws_size,
                              hipStream_t stream) {
  const float* x  = (const float*)d_in[0];
  const float* Wq = (const float*)d_in[1];
  const float* Wk = (const float*)d_in[2];
  const float* Wv = (const float*)d_in[3];
  const float* Wo = (const float*)d_in[4];

  char* ws = (char*)d_ws;
  constexpr size_t SZ_X = (size_t)M * K * 2;   // 16 MiB
  constexpr size_t SZ_W = (size_t)N * K * 2;   //  2 MiB
  constexpr size_t SZ_P = (size_t)M * N * 2;   // 16 MiB (Q/K/Vt/O each)
  bf16* xb  = (bf16*)(ws);
  bf16* wqb = (bf16*)(ws + SZ_X);
  bf16* wkb = (bf16*)(ws + SZ_X + SZ_W);
  bf16* wvb = (bf16*)(ws + SZ_X + 2 * SZ_W);
  bf16* wob = (bf16*)(ws + SZ_X + 3 * SZ_W);
  bf16* Qb  = (bf16*)(ws + SZ_X + 4 * SZ_W);
  bf16* Kb  = (bf16*)(ws + SZ_X + 4 * SZ_W + SZ_P);
  bf16* Vtb = (bf16*)(ws + SZ_X + 4 * SZ_W + 2 * SZ_P);
  bf16* Ob  = (bf16*)(ws + SZ_X + 4 * SZ_W + 3 * SZ_P);

  cvt_bf16<<<(M * K) / 1024, 256, 0, stream>>>(x, xb);
  cvt_bf16<<<(N * K) / 1024, 256, 0, stream>>>(Wq, wqb);
  cvt_bf16<<<(N * K) / 1024, 256, 0, stream>>>(Wk, wkb);
  cvt_bf16<<<(N * K) / 1024, 256, 0, stream>>>(Wv, wvb);
  cvt_bf16<<<(N * K) / 1024, 256, 0, stream>>>(Wo, wob);

  dim3 ggrid(M / 128, N / 128);
  gemm_bf16<1><<<ggrid, 256, 0, stream>>>(xb, wqb, Qb, 0.125f);  // Q, pre-scaled
  gemm_bf16<1><<<ggrid, 256, 0, stream>>>(xb, wkb, Kb, 1.0f);
  gemm_bf16<2><<<ggrid, 256, 0, stream>>>(xb, wvb, Vtb, 1.0f);

  attn<<<dim3(T / 128, Bb * H), 256, 0, stream>>>(Qb, Kb, Vtb, Ob);

  gemm_bf16<0><<<ggrid, 256, 0, stream>>>(Ob, wob, d_out, 1.0f);
}
